// GATBlock_67173288509940
// MI455X (gfx1250) — matrix operand order repaired and checked by reading, not compiled, not run on hardware
//
#include <hip/hip_runtime.h>
#include <hip/hip_bf16.h>

// ---------------------------------------------------------------------------
// Problem constants (from reference)
// ---------------------------------------------------------------------------
#define NN      50000
#define EE      800000
#define E2      (EE + NN)       // with self loops
#define HH      4
#define CC      64              // per-head channels (both layers)
#define FOUT    256             // H*C, both layers
#define FIN     128
#define EDGE_D  16
#define NEG_ATT 0.2f
#define NEG_OUT 0.1f
#define BN_EPS  1e-5f

typedef __attribute__((ext_vector_type(16))) __bf16          v16bf;
typedef __attribute__((ext_vector_type(8)))  float           v8f;
typedef __attribute__((ext_vector_type(8)))  unsigned short  us8;
typedef __attribute__((ext_vector_type(16))) unsigned short  us16;
typedef __attribute__((ext_vector_type(4)))  unsigned int    u32x4;
typedef __attribute__((ext_vector_type(8)))  int             i32x8;
typedef __attribute__((ext_vector_type(4)))  int             i32x4;

// ---------------------------------------------------------------------------
// helpers
// ---------------------------------------------------------------------------
__device__ __forceinline__ unsigned short f2bf(float f) {
    unsigned int u = __float_as_uint(f);
    u = u + 0x7FFFu + ((u >> 16) & 1u);   // round to nearest even
    return (unsigned short)(u >> 16);
}

__device__ __forceinline__ void atomicMaxF(float* addr, float v) {
    if (v >= 0.0f) atomicMax((int*)addr, __float_as_int(v));
    else           atomicMin((unsigned int*)addr, (unsigned int)__float_as_int(v));
}

__global__ void k_fill_f32(float* __restrict__ p, float v, long long n) {
    long long i = (long long)blockIdx.x * blockDim.x + threadIdx.x;
    if (i < n) p[i] = v;
}

// W[K,Nc] f32 -> Wt[Nc,K] bf16 (so B fragments become contiguous b128 loads)
__global__ void k_transpose_bf16(const float* __restrict__ W, unsigned short* __restrict__ Wt,
                                 int K, int Nc) {
    int i = blockIdx.x * blockDim.x + threadIdx.x;
    if (i >= K * Nc) return;
    int k = i / Nc, n = i - k * Nc;
    Wt[(size_t)n * K + k] = f2bf(W[i]);
}

// ---------------------------------------------------------------------------
// BatchNorm: per-column mean / rstd over node dimension
// ---------------------------------------------------------------------------
__global__ void k_bn_stats(const float* __restrict__ x, float* __restrict__ meanO,
                           float* __restrict__ rstdO, int rows, int ld) {
    __shared__ float ssum[256];
    __shared__ float ssq[256];
    int col = blockIdx.x;
    float s = 0.f, q = 0.f;
    for (int r = threadIdx.x; r < rows; r += blockDim.x) {
        float v = x[(size_t)r * ld + col];
        s += v; q += v * v;
    }
    ssum[threadIdx.x] = s; ssq[threadIdx.x] = q;
    __syncthreads();
    for (int off = 128; off > 0; off >>= 1) {
        if ((int)threadIdx.x < off) {
            ssum[threadIdx.x] += ssum[threadIdx.x + off];
            ssq[threadIdx.x]  += ssq[threadIdx.x + off];
        }
        __syncthreads();
    }
    if (threadIdx.x == 0) {
        float m  = ssum[0] / (float)rows;
        float vr = ssq[0] / (float)rows - m * m;
        meanO[col] = m;
        rstdO[col] = rsqrtf(vr + BN_EPS);
    }
}

// BN apply + leaky_relu(0.1) + cast to bf16 (GEMM A operand)
__global__ void k_bn_apply(const float* __restrict__ x, const float* __restrict__ mean,
                           const float* __restrict__ rstd, const float* __restrict__ gamma,
                           const float* __restrict__ beta, unsigned short* __restrict__ xb,
                           int total, int ld) {
    int i = blockIdx.x * blockDim.x + threadIdx.x;
    if (i >= total) return;
    int col = i & (ld - 1);   // ld is 128 or 256 (power of two)
    float v = (x[i] - mean[col]) * rstd[col] * gamma[col] + beta[col];
    v = v > 0.f ? v : NEG_OUT * v;
    xb[i] = f2bf(v);
}

// ---------------------------------------------------------------------------
// WMMA bf16 GEMM:  C[M,Nc] = A[M,K] * Wt[Nc,K]^T   (fp32 accumulate)
//
// Grid: x = 8-mtile strips, y = 64-column group.  All 8 waves of a block share
// one 64xK weight panel, staged into LDS once by wave 0 via the Tensor Data
// Mover (tensor_load_to_lds + s_wait_tensorcnt), then consumed with
// ds_load_b128 fragments.  A is streamed with global_load_b128 + prefetch.
//
// Fragment layouts (wave32):
//   A (16-bit 16x32): lanes 0-15 row=lane  K=0..15;
//                     lanes 16-31 row=lane-16, K=16..31
//   B (16-bit 32x16): lanes 0-15 col=lane K=0..15; lanes 16-31 col=lane-16 K=16..31
//     -> with Wt[N,K] panel in LDS, a B fragment is 16 contiguous bf16 (32B aligned)
//   C/D: VGPR r -> row r (lanes 0-15) / row r+8 (lanes 16-31), col = lane&15
// M%16==0, K%32==0 (K<=256), Nc%64==0 guaranteed here.
// ---------------------------------------------------------------------------
__global__ void k_wmma_gemm_bf16(const unsigned short* __restrict__ A,
                                 const unsigned short* __restrict__ Bt,
                                 float* __restrict__ C,
                                 int M, int K, int Nc) {
    __shared__ unsigned short shB[64 * 256];     // 32 KB: 64 cols x K(<=256) bf16

    const int waveId = threadIdx.x >> 5;
    const int lane   = threadIdx.x & 31;
    const int g4     = blockIdx.y;               // 64-column group
    const int mtile  = blockIdx.x * 8 + waveId;
    const int ncBase = g4 * 64;

    // ---- TDM: stage the 64xK weight panel into LDS (wave 0 only) ----
    if (waveId == 0) {
        unsigned long long ga = (unsigned long long)(const void*)(Bt + (size_t)ncBase * K);
        unsigned int ldsOff   = (unsigned int)(unsigned long long)(const void*)&shB[0];
        u32x4 g0;
        g0[0] = 1u;                                            // count=1, user descriptor
        g0[1] = ldsOff;                                        // lds_addr
        g0[2] = (unsigned int)(ga & 0xFFFFFFFFu);              // global_addr[31:0]
        g0[3] = (unsigned int)((ga >> 32) & 0x1FFFFFFu) | (2u << 30);  // addr[56:32] | type=2
        i32x8 g1;
        g1[0] = (int)(1u << 16);                               // data_size=1 (2 bytes)
        g1[1] = (int)(((unsigned)K & 0xFFFFu) << 16);          // tensor_dim0 lo16 @bit48
        g1[2] = (int)((((unsigned)K >> 16) & 0xFFFFu) | (64u << 16)); // dim0 hi | tensor_dim1 lo
        g1[3] = (int)(((unsigned)K & 0xFFFFu) << 16);          // tensor_dim1 hi(0) | tile_dim0=K
        g1[4] = 64;                                            // tile_dim1=64, tile_dim2=0
        g1[5] = K;                                             // tensor_dim0_stride lo32
        g1[6] = 0;                                             // stride hi | dim1_stride lo
        g1[7] = 0;
        i32x4 gz = {0, 0, 0, 0};
#if defined(__clang_major__) && (__clang_major__ >= 23)
        i32x8 gz8 = {0, 0, 0, 0, 0, 0, 0, 0};
        __builtin_amdgcn_tensor_load_to_lds(g0, g1, gz, gz, gz8, 0);
#else
        __builtin_amdgcn_tensor_load_to_lds(g0, g1, gz, gz, 0);
#endif
        __builtin_amdgcn_s_wait_tensorcnt((short)0);
    }
    __syncthreads();   // publish LDS panel to all waves (every wave reaches this)

    if (mtile * 16 < M) {
        const int rowA   = lane & 15;
        const int kbaseA = (lane < 16) ? 0 : 16;
        const int colB   = lane & 15;
        const int kbaseB = (lane < 16) ? 0 : 16;

        const unsigned short* Arow = A + (size_t)(mtile * 16 + rowA) * K;

        v8f acc0 = {}, acc1 = {}, acc2 = {}, acc3 = {};

        for (int k0 = 0; k0 < K; k0 += 32) {
            // A fragment: two aligned 16B global loads
            us8 alo = *(const us8*)(Arow + k0 + kbaseA);
            us8 ahi = *(const us8*)(Arow + k0 + 8 + kbaseA);
            us16 au;
#pragma unroll
            for (int j = 0; j < 8; ++j) { au[j] = alo[j]; au[8 + j] = ahi[j]; }
            v16bf a = __builtin_bit_cast(v16bf, au);

            if (k0 + 32 < K) __builtin_prefetch(Arow + k0 + 32, 0, 1);

            // 4 B fragments from LDS: one aligned 32B ds read each
            const int kOff = k0 + kbaseB;
            v16bf b0 = __builtin_bit_cast(v16bf, *(const us16*)(shB + ( 0 + colB) * K + kOff));
            v16bf b1 = __builtin_bit_cast(v16bf, *(const us16*)(shB + (16 + colB) * K + kOff));
            v16bf b2 = __builtin_bit_cast(v16bf, *(const us16*)(shB + (32 + colB) * K + kOff));
            v16bf b3 = __builtin_bit_cast(v16bf, *(const us16*)(shB + (48 + colB) * K + kOff));

            acc0 = __builtin_amdgcn_wmma_f32_16x16x32_bf16(false, a, false, b0, (short)0, acc0, false, false);
            acc1 = __builtin_amdgcn_wmma_f32_16x16x32_bf16(false, a, false, b1, (short)0, acc1, false, false);
            acc2 = __builtin_amdgcn_wmma_f32_16x16x32_bf16(false, a, false, b2, (short)0, acc2, false, false);
            acc3 = __builtin_amdgcn_wmma_f32_16x16x32_bf16(false, a, false, b3, (short)0, acc3, false, false);
        }

        const int rowBase = mtile * 16 + ((lane < 16) ? 0 : 8);
        float* Cb = C + (size_t)rowBase * Nc + ncBase + colB;
#pragma unroll
        for (int r = 0; r < 8; ++r) {
            float* Cr = Cb + (size_t)r * Nc;
            Cr[ 0] = acc0[r];
            Cr[16] = acc1[r];
            Cr[32] = acc2[r];
            Cr[48] = acc3[r];
        }
    }
}

// ---------------------------------------------------------------------------
// Attention pieces
// ---------------------------------------------------------------------------
// e_mean: column means of edge_attr [E,16]
__global__ void k_edge_mean(const float* __restrict__ ea, float* __restrict__ outm, int rows) {
    __shared__ float ssum[256];
    int col = blockIdx.x;
    float s = 0.f;
    for (int r = threadIdx.x; r < rows; r += blockDim.x) s += ea[(size_t)r * EDGE_D + col];
    ssum[threadIdx.x] = s;
    __syncthreads();
    for (int off = 128; off > 0; off >>= 1) {
        if ((int)threadIdx.x < off) ssum[threadIdx.x] += ssum[threadIdx.x + off];
        __syncthreads();
    }
    if (threadIdx.x == 0) outm[col] = ssum[0] / (float)rows;
}

// weff[k][h] = sum_c We[k, h*64+c] * a_edge[h*64+c]   (collapses eattr@We to 16x4)
__global__ void k_weff(const float* __restrict__ We, const float* __restrict__ aedge,
                       float* __restrict__ weff) {
    int t = threadIdx.x;
    if (t >= EDGE_D * HH) return;
    int k = t >> 2, h = t & 3;
    float s = 0.f;
#pragma unroll
    for (int c = 0; c < CC; ++c) s += We[k * FOUT + h * CC + c] * aedge[h * CC + c];
    weff[k * HH + h] = s;
}

// per-node attention dots: alsrc[n,h] = h[n,h,:]·a_src[h,:], aldst likewise
__global__ void k_node_al(const float* __restrict__ hbuf, const float* __restrict__ asrc,
                          const float* __restrict__ adst, float* __restrict__ alsrc,
                          float* __restrict__ aldst) {
    int t = blockIdx.x * blockDim.x + threadIdx.x;
    if (t >= NN * HH) return;
    int n = t >> 2, h = t & 3;
    const float* hp = hbuf + (size_t)n * FOUT + h * CC;
    const float* as = asrc + h * CC;
    const float* ad = adst + h * CC;
    float s = 0.f, d = 0.f;
#pragma unroll
    for (int c = 0; c < CC; ++c) { s += hp[c] * as[c]; d += hp[c] * ad[c]; }
    alsrc[t] = s; aldst[t] = d;
}

// logits = leaky_relu(al_src[src] + al_dst[dst] + eattr·weff, 0.2); atomic seg-max into m
__global__ void k_att_logits(const float* __restrict__ alsrc, const float* __restrict__ aldst,
                             const float* __restrict__ eattr, const float* __restrict__ emean,
                             const float* __restrict__ weff, const int* __restrict__ ei,
                             float* __restrict__ logits, float* __restrict__ mbuf) {
    int e = blockIdx.x * blockDim.x + threadIdx.x;
    if (e >= E2) return;
    int s, d;
    float attr[EDGE_D];
    if (e < EE) {
        s = ei[e]; d = ei[EE + e];
#pragma unroll
        for (int k = 0; k < EDGE_D; ++k) attr[k] = eattr[(size_t)e * EDGE_D + k];
    } else {
        s = d = e - EE;
#pragma unroll
        for (int k = 0; k < EDGE_D; ++k) attr[k] = emean[k];
    }
    float acc[HH] = {0.f, 0.f, 0.f, 0.f};
#pragma unroll
    for (int k = 0; k < EDGE_D; ++k) {
        float av = attr[k];
#pragma unroll
        for (int h = 0; h < HH; ++h) acc[h] += av * weff[k * HH + h];
    }
    float4 as4 = ((const float4*)alsrc)[s];
    float4 ad4 = ((const float4*)aldst)[d];
    float sv[HH] = {as4.x + ad4.x, as4.y + ad4.y, as4.z + ad4.z, as4.w + ad4.w};
#pragma unroll
    for (int h = 0; h < HH; ++h) {
        float v = sv[h] + acc[h];
        v = v > 0.f ? v : NEG_ATT * v;
        logits[(size_t)e * HH + h] = v;
        atomicMaxF(&mbuf[d * HH + h], v);
    }
}

// ex = exp(logit - m[dst]); store in place; atomic seg-sum into denom
__global__ void k_att_expsum(const int* __restrict__ ei, float* __restrict__ logits,
                             const float* __restrict__ mbuf, float* __restrict__ denom) {
    int e = blockIdx.x * blockDim.x + threadIdx.x;
    if (e >= E2) return;
    int d = (e < EE) ? ei[EE + e] : (e - EE);
    float4 m4 = ((const float4*)mbuf)[d];
    float mv[HH] = {m4.x, m4.y, m4.z, m4.w};
#pragma unroll
    for (int h = 0; h < HH; ++h) {
        float ex = __expf(logits[(size_t)e * HH + h] - mv[h]);
        logits[(size_t)e * HH + h] = ex;
        atomicAdd(&denom[d * HH + h], ex);
    }
}

// out[dst] += alpha * h[src]  (one thread per (edge,head); out lives in 192MB L2)
__global__ void k_att_aggregate(const int* __restrict__ ei, const float* __restrict__ ex,
                                const float* __restrict__ denom, const float* __restrict__ hbuf,
                                float* __restrict__ out) {
    int t = blockIdx.x * blockDim.x + threadIdx.x;
    if (t >= E2 * HH) return;
    int e = t >> 2, h = t & 3;
    int s, d;
    if (e < EE) { s = ei[e]; d = ei[EE + e]; } else { s = d = e - EE; }
    float alpha = ex[(size_t)e * HH + h] / denom[d * HH + h];
    const float4* hs = (const float4*)(hbuf + (size_t)s * FOUT + h * CC);
    float* op = out + (size_t)d * FOUT + h * CC;
#pragma unroll
    for (int c4 = 0; c4 < CC / 4; ++c4) {
        float4 v = hs[c4];
        atomicAdd(op + c4 * 4 + 0, v.x * alpha);
        atomicAdd(op + c4 * 4 + 1, v.y * alpha);
        atomicAdd(op + c4 * 4 + 2, v.z * alpha);
        atomicAdd(op + c4 * 4 + 3, v.w * alpha);
    }
}

// x = leaky_relu(x + bias, 0.1) in place
__global__ void k_bias_lrelu(float* __restrict__ x, const float* __restrict__ b, int total) {
    int i = blockIdx.x * blockDim.x + threadIdx.x;
    if (i >= total) return;
    float v = x[i] + b[i & (FOUT - 1)];
    x[i] = v > 0.f ? v : NEG_OUT * v;
}

// ---------------------------------------------------------------------------
// Host launcher
// ---------------------------------------------------------------------------
static inline size_t alignUp(size_t x) { return (x + 255) & ~(size_t)255; }

extern "C" void kernel_launch(void* const* d_in, const int* in_sizes, int n_in,
                              void* d_out, int out_size, void* d_ws, size_t ws_size,
                              hipStream_t stream) {
    const float* x       = (const float*)d_in[0];
    const int*   ei      = (const int*)  d_in[1];
    const float* eattr   = (const float*)d_in[2];
    const float* bn1g    = (const float*)d_in[3];
    const float* bn1b    = (const float*)d_in[4];
    const float* W1      = (const float*)d_in[5];
    const float* We1     = (const float*)d_in[6];
    const float* asrc1   = (const float*)d_in[7];
    const float* adst1   = (const float*)d_in[8];
    const float* aedge1  = (const float*)d_in[9];
    const float* b1      = (const float*)d_in[10];
    const float* bn2g    = (const float*)d_in[11];
    const float* bn2b    = (const float*)d_in[12];
    const float* W2      = (const float*)d_in[13];
    const float* We2     = (const float*)d_in[14];
    const float* asrc2   = (const float*)d_in[15];
    const float* adst2   = (const float*)d_in[16];
    const float* aedge2  = (const float*)d_in[17];
    const float* b2      = (const float*)d_in[18];
    float* out = (float*)d_out;

    // ---- workspace carve ----
    char* ws = (char*)d_ws;
    size_t off = 0;
    float* hbuf  = (float*)(ws + off); off = alignUp(off + (size_t)NN * FOUT * 4);   // 51.2 MB
    float* agg   = (float*)(ws + off); off = alignUp(off + (size_t)NN * FOUT * 4);   // 51.2 MB
    unsigned short* xb = (unsigned short*)(ws + off); off = alignUp(off + (size_t)NN * FOUT * 2); // 25.6 MB
    float* logit = (float*)(ws + off); off = alignUp(off + (size_t)E2 * HH * 4);     // 13.6 MB
    float* mbuf  = (float*)(ws + off); off = alignUp(off + (size_t)NN * HH * 4);
    float* denom = (float*)(ws + off); off = alignUp(off + (size_t)NN * HH * 4);
    float* alsrc = (float*)(ws + off); off = alignUp(off + (size_t)NN * HH * 4);
    float* aldst = (float*)(ws + off); off = alignUp(off + (size_t)NN * HH * 4);
    unsigned short* Wt = (unsigned short*)(ws + off); off = alignUp(off + (size_t)FOUT * FOUT * 2);
    float* emean = (float*)(ws + off); off = alignUp(off + EDGE_D * 4);
    float* weff  = (float*)(ws + off); off = alignUp(off + EDGE_D * HH * 4);
    float* bnm   = (float*)(ws + off); off = alignUp(off + FOUT * 4);
    float* bnr   = (float*)(ws + off); off = alignUp(off + FOUT * 4);
    (void)ws_size; (void)n_in; (void)in_sizes; (void)out_size;

    const int B = 256;
    const long long NF = (long long)NN * FOUT;
    const int gNF = (int)((NF + B - 1) / B);
    const int gNH = (NN * HH + B - 1) / B;
    const int gE2 = (E2 + B - 1) / B;
    const int gEH = (E2 * HH + B - 1) / B;
    // GEMM grid: x = 8-mtile strips, y = 64-column groups
    const dim3 gemmGrid((NN / 16 + 7) / 8, FOUT / 64);

    // shared: e_mean of edge_attr (used by both layers)
    k_edge_mean<<<EDGE_D, B, 0, stream>>>(eattr, emean, EE);

    // ======================= Layer 1 =======================
    k_bn_stats<<<FIN, B, 0, stream>>>(x, bnm, bnr, NN, FIN);
    k_bn_apply<<<(NN * FIN + B - 1) / B, B, 0, stream>>>(x, bnm, bnr, bn1g, bn1b, xb, NN * FIN, FIN);
    k_transpose_bf16<<<(FIN * FOUT + B - 1) / B, B, 0, stream>>>(W1, Wt, FIN, FOUT);
    k_wmma_gemm_bf16<<<gemmGrid, B, 0, stream>>>(xb, Wt, hbuf, NN, FIN, FOUT);

    k_node_al<<<gNH, B, 0, stream>>>(hbuf, asrc1, adst1, alsrc, aldst);
    k_weff<<<1, 64, 0, stream>>>(We1, aedge1, weff);

    k_fill_f32<<<gNH, B, 0, stream>>>(mbuf, -3.402823466e38f, (long long)NN * HH);
    k_fill_f32<<<gNH, B, 0, stream>>>(denom, 0.f, (long long)NN * HH);
    k_fill_f32<<<gNF, B, 0, stream>>>(agg, 0.f, NF);

    k_att_logits<<<gE2, B, 0, stream>>>(alsrc, aldst, eattr, emean, weff, ei, logit, mbuf);
    k_att_expsum<<<gE2, B, 0, stream>>>(ei, logit, mbuf, denom);
    k_att_aggregate<<<gEH, B, 0, stream>>>(ei, logit, denom, hbuf, agg);
    k_bias_lrelu<<<gNF, B, 0, stream>>>(agg, b1, (int)NF);

    // ======================= Layer 2 =======================
    k_bn_stats<<<FOUT, B, 0, stream>>>(agg, bnm, bnr, NN, FOUT);
    k_bn_apply<<<gNF, B, 0, stream>>>(agg, bnm, bnr, bn2g, bn2b, xb, (int)NF, FOUT);
    k_transpose_bf16<<<(FOUT * FOUT + B - 1) / B, B, 0, stream>>>(W2, Wt, FOUT, FOUT);
    k_wmma_gemm_bf16<<<gemmGrid, B, 0, stream>>>(xb, Wt, hbuf, NN, FOUT, FOUT);

    k_node_al<<<gNH, B, 0, stream>>>(hbuf, asrc2, adst2, alsrc, aldst);
    k_weff<<<1, 64, 0, stream>>>(We2, aedge2, weff);

    k_fill_f32<<<gNH, B, 0, stream>>>(mbuf, -3.402823466e38f, (long long)NN * HH);
    k_fill_f32<<<gNH, B, 0, stream>>>(denom, 0.f, (long long)NN * HH);
    k_fill_f32<<<gNF, B, 0, stream>>>(out, 0.f, NF);

    k_att_logits<<<gE2, B, 0, stream>>>(alsrc, aldst, eattr, emean, weff, ei, logit, mbuf);
    k_att_expsum<<<gE2, B, 0, stream>>>(ei, logit, mbuf, denom);
    k_att_aggregate<<<gEH, B, 0, stream>>>(ei, logit, denom, hbuf, out);
    k_bias_lrelu<<<gNF, B, 0, stream>>>(out, b2, (int)NF);

    // second output: edge_index bits appended after the node features
    hipMemcpyAsync(out + NF, ei, (size_t)2 * EE * sizeof(int),
                   hipMemcpyDeviceToDevice, stream);
}
